// PRISMLayer_63960652972058
// MI455X (gfx1250) — compile-verified, matrix operand order checked by reading, not run on hardware
//
#include <hip/hip_runtime.h>
#include <cmath>

// Problem constants (match reference)
#define B_ 4
#define T_ 4096
#define D_ 1024
#define C_ 8
#define DC_ 128
#define MLP_ 2048
#define M_ (B_*T_)   // 16384 token rows

typedef _Float16 h16;
typedef __attribute__((ext_vector_type(8)))  _Float16 v8h;
typedef __attribute__((ext_vector_type(16))) _Float16 v16h;
typedef __attribute__((ext_vector_type(8)))  float    v8f;
typedef unsigned int u32;
typedef __attribute__((ext_vector_type(4))) u32 u32x4;
typedef __attribute__((ext_vector_type(4))) int i32x4;
typedef __attribute__((ext_vector_type(8))) int i32x8;

// LDS tile: 64 rows x 64 halves, padded to 72-half row stride (TDM pad feature)
#define SB_STRIDE 72

// ---------------------------------------------------------------------------
// WMMA fragment helpers (CDNA5: V_WMMA_F32_16X16X32_F16, wave32)
// lanes 0-15 carry K = {0..7, 16..23}; lanes 16-31 carry K = {8..15, 24..31}
// ---------------------------------------------------------------------------
static __device__ __forceinline__ v16h load_frag(const h16* __restrict__ base,
                                                 int ld, int r0, int k0, int lane) {
  const int m    = r0 + (lane & 15);
  const int koff = (lane >> 4) << 3;   // 0 or 8
  const h16* p = base + (size_t)m * ld + k0 + koff;
  v8h lo = *(const v8h*)(p);
  v8h hi = *(const v8h*)(p + 16);
  return __builtin_shufflevector(lo, hi, 0,1,2,3,4,5,6,7,8,9,10,11,12,13,14,15);
}

static __device__ __forceinline__ v16h lds_frag(const h16* sb, int n0, int kk, int lane) {
  const int n    = n0 + (lane & 15);
  const int koff = (lane >> 4) << 3;
  const h16* p = sb + n * SB_STRIDE + kk + koff;
  v8h lo = *(const v8h*)(p);
  v8h hi = *(const v8h*)(p + 16);
  return __builtin_shufflevector(lo, hi, 0,1,2,3,4,5,6,7,8,9,10,11,12,13,14,15);
}

static __device__ __forceinline__ v8f wmma32(v16h a, v16h b, v8f c) {
  return __builtin_amdgcn_wmma_f32_16x16x32_f16(false, a, false, b, (short)0, c,
                                                false, false);
}

// ---------------------------------------------------------------------------
// TDM: DMA a 64x64 f16 tile (row stride K elems) from global into LDS,
// padding 4 DWORDs after every 32 DWORDs (one row) -> 72-half LDS row stride.
// D# layout per CDNA5 ISA.
// 6-arg clang-23 builtin: (uint32x4 g0, int32x8 g1, int32x4, int32x4,
//                          int32x8, i32 cpol)
// ---------------------------------------------------------------------------
static __device__ __forceinline__ void tdm_load_tile(const h16* gsrc,
                                                     u32 ldsByteAddr, int K) {
  const unsigned long long ga = (unsigned long long)(uintptr_t)gsrc;
  u32x4 g0;
  g0.x = 1u;                                            // count=1, user mode
  g0.y = ldsByteAddr;                                   // lds_addr (bytes)
  g0.z = (u32)ga;                                       // global_addr[31:0]
  g0.w = (u32)((ga >> 32) & 0x1FFFFFFull) | 0x80000000u; // addr[56:32] | type=2
  i32x8 g1;
  g1[0] = (1 << 16) | (1 << 20) | (4 << 22) | (3 << 25); // 2B elems, pad_en,
                                                         // interval=32dw, pad=4dw
  g1[1] = (K & 0xFFFF) << 16;                    // tensor_dim0 lo16
  g1[2] = ((K >> 16) & 0xFFFF) | (64 << 16);     // tensor_dim0 hi | tensor_dim1=64
  g1[3] = (64 << 16);                            // tensor_dim1 hi | tile_dim0=64
  g1[4] = 64;                                    // tile_dim1=64, tile_dim2=0
  g1[5] = K;                                     // tensor_dim0_stride lo32
  g1[6] = 0;
  g1[7] = 0;
  i32x4 z4 = {};
  i32x8 z8 = {};
  __builtin_amdgcn_tensor_load_to_lds(g0, g1, z4, z4, z8, 0);
}

// ---------------------------------------------------------------------------
// Generic NT GEMM: out[M,N] = A[M,K](f16) * W[N,K](f16)^T  (+bias, +epilogue)
// Block: 256 thr = 8 waves; block tile 128(M) x 64(N); wave tile 16 x 64.
// Weight strip (64 x K) staged through LDS by TDM, double-buffered in 64-col
// stages; A fragments software-pipelined one stage ahead; all 16 ds_loads of
// a stage's B fragments are issued before the 8-WMMA burst so dscnt waits
// overlap with matrix work.
// EPI: 0 f32 | 1 f32+f16 | 2 sigmoid(acc+b)*aux | 3 sigmoid(acc+b)
//      4 gelu(acc+b)->f16 | 5 aux+acc+b -> f32
// ---------------------------------------------------------------------------
template<int EPI>
__global__ __launch_bounds__(256) void gemm_epi(
    const h16*  __restrict__ A,    size_t strideA,
    const h16*  __restrict__ W,    size_t strideW,
    const float* __restrict__ bias, size_t strideBias,
    const float* __restrict__ aux,  size_t strideAux,
    float* __restrict__ outF, size_t strideOutF,
    h16*   __restrict__ outH, size_t strideOutH,
    int M, int N, int K)
{
  __shared__ h16 sB[2][64 * SB_STRIDE];

  const int c = blockIdx.z;
  A += (size_t)c * strideA;
  W += (size_t)c * strideW;
  if constexpr (EPI >= 2)             bias += (size_t)c * strideBias;
  if constexpr (EPI == 2 || EPI == 5) aux  += (size_t)c * strideAux;

  const int lane = threadIdx.x & 31;
  const int wave = threadIdx.x >> 5;
  const bool w0  = (wave == 0);
  const int row0 = blockIdx.y * 128 + wave * 16;
  const int col0 = blockIdx.x * 64;
  const h16* Wblk = W + (size_t)col0 * K;

  // Prologue: stage 0 weight tile via TDM; preload stage-0 A fragments.
  if (w0) tdm_load_tile(Wblk, (u32)(uintptr_t)&sB[0][0], K);
  v16h a0 = load_frag(A, K, row0, 0, lane);
  v16h a1 = load_frag(A, K, row0, 32, lane);
  if (w0) __builtin_amdgcn_s_wait_tensorcnt(0);
  __syncthreads();

  v8f acc[4] = {};
  int buf = 0;
  for (int k0 = 0; k0 < K; k0 += 64) {
    const int kn = k0 + 64;
    // Issue next stage's DMA into the other buffer while we compute.
    if (kn < K && w0)
      tdm_load_tile(Wblk + kn, (u32)(uintptr_t)&sB[buf ^ 1][0], K);
    // Pipeline next stage's A fragments ahead of this stage's WMMAs.
    v16h a0n = a0, a1n = a1;
    if (kn < K) {
      a0n = load_frag(A, K, row0, kn, lane);
      a1n = load_frag(A, K, row0, kn + 32, lane);
    }
    const h16* sb = &sB[buf][0];
    // Batch all B-fragment LDS reads for this stage, then burst the WMMAs.
    v16h bfr[8];
    #pragma unroll
    for (int t = 0; t < 4; ++t) bfr[t]     = lds_frag(sb, t * 16, 0,  lane);
    #pragma unroll
    for (int t = 0; t < 4; ++t) bfr[4 + t] = lds_frag(sb, t * 16, 32, lane);
    #pragma unroll
    for (int t = 0; t < 4; ++t) acc[t] = wmma32(a0, bfr[t],     acc[t]);
    #pragma unroll
    for (int t = 0; t < 4; ++t) acc[t] = wmma32(a1, bfr[4 + t], acc[t]);
    if (kn < K) {
      if (w0) __builtin_amdgcn_s_wait_tensorcnt(0); // next tile landed
      __syncthreads();                              // publish + retire reads
      buf ^= 1;
      a0 = a0n; a1 = a1n;
    }
  }

  // D layout: VGPR r -> M = r + (lane>=16 ? 8 : 0); N = lane & 15
  const int mbase = row0 + ((lane >> 4) << 3);
  const int nb    = col0 + (lane & 15);
  #pragma unroll
  for (int t = 0; t < 4; ++t) {
    const int n = nb + t * 16;
    float bv = 0.f;
    if constexpr (EPI >= 2) bv = bias[n];
    #pragma unroll
    for (int r = 0; r < 8; ++r) {
      const size_t o = (size_t)(mbase + r) * N + n;
      const float v = acc[t][r];
      if constexpr (EPI == 0) {
        outF[(size_t)c * strideOutF + o] = v;
      } else if constexpr (EPI == 1) {
        outF[(size_t)c * strideOutF + o] = v;
        outH[(size_t)c * strideOutH + o] = (h16)v;
      } else if constexpr (EPI == 2) {
        const float s = 1.f / (1.f + __expf(-(v + bv)));
        outF[(size_t)c * strideOutF + o] = s * aux[o];
      } else if constexpr (EPI == 3) {
        outF[o] = 1.f / (1.f + __expf(-(v + bv)));
      } else if constexpr (EPI == 4) {
        const float xg = v + bv;
        outH[o] = (h16)(0.5f * xg * (1.f + erff(xg * 0.70710678118f)));
      } else {
        outF[o] = aux[o] + v + bv;
      }
    }
  }
}

// ---------------------------------------------------------------------------
// Elementwise / support kernels
// ---------------------------------------------------------------------------
__global__ void cvt_kernel(const float* __restrict__ s, h16* __restrict__ d, size_t n) {
  size_t i = (size_t)blockIdx.x * blockDim.x + threadIdx.x;
  const size_t stride = (size_t)gridDim.x * blockDim.x;
  for (; i < n; i += stride) d[i] = (h16)s[i];
}

// (C,DC,DC) [c][i][j] f32 -> [c][j][i] f16  (so weights become W[n][k])
__global__ void cvtT_kernel(const float* __restrict__ s, h16* __restrict__ d) {
  const int c = blockIdx.x / DC_, j = blockIdx.x % DC_, i = threadIdx.x;
  d[(size_t)blockIdx.x * DC_ + i] = (h16)s[((size_t)c * DC_ + i) * DC_ + j];
}

__global__ __launch_bounds__(256) void ln_kernel(const float* __restrict__ x,
                                                 const float* __restrict__ g,
                                                 const float* __restrict__ b,
                                                 h16* __restrict__ out) {
  __shared__ float red[256];
  const int row = blockIdx.x, tid = threadIdx.x;
  const float* xr = x + (size_t)row * D_;
  float s = 0.f;
  for (int i = tid; i < D_; i += 256) s += xr[i];
  red[tid] = s; __syncthreads();
  for (int o = 128; o > 0; o >>= 1) { if (tid < o) red[tid] += red[tid + o]; __syncthreads(); }
  const float mu = red[0] * (1.f / D_);
  __syncthreads();
  float q = 0.f;
  for (int i = tid; i < D_; i += 256) { const float dd = xr[i] - mu; q += dd * dd; }
  red[tid] = q; __syncthreads();
  for (int o = 128; o > 0; o >>= 1) { if (tid < o) red[tid] += red[tid + o]; __syncthreads(); }
  const float rstd = rsqrtf(red[0] * (1.f / D_) + 1e-5f);
  h16* orow = out + (size_t)row * D_;
  for (int i = tid; i < D_; i += 256)
    orow[i] = (h16)((xr[i] - mu) * rstd * g[i] + b[i]);
}

// Linear recurrence h_t = v_t + lam_c * h_{t-+1}; one thread per (c,b,dc) lane.
__global__ void scan_kernel(const float* __restrict__ v, float* __restrict__ H,
                            h16* __restrict__ Hh, int backward) {
  const int cb = blockIdx.x;
  const int c = cb / B_, b = cb % B_;
  const int dc = threadIdx.x;
  const float lam = 1.f - exp2f(-(13.0f / 7.0f) * (float)c); // log2(8192)/(C-1)
  const size_t base = ((size_t)c * M_ + (size_t)b * T_) * DC_ + dc;
  float h = 0.f;
  if (!backward) {
    for (int t = 0; t < T_; ++t) {
      const size_t i = base + (size_t)t * DC_;
      h = v[i] + lam * h; H[i] = h; Hh[i] = (h16)h;
    }
  } else {
    for (int t = T_ - 1; t >= 0; --t) {
      const size_t i = base + (size_t)t * DC_;
      h = v[i] + lam * h; H[i] = h; Hh[i] = (h16)h;
    }
  }
}

// M = H + UH * (alpha*(1-I) @ VH); writes f32 (b,t,c*DC+dc) + f16 into fuse input
__global__ __launch_bounds__(256) void interfere_kernel(
    const float* __restrict__ H, const float* __restrict__ UH,
    const float* __restrict__ VH, const float* __restrict__ alpha,
    float* __restrict__ mOut, h16* __restrict__ fuseH, int colBase) {
  __shared__ float a[64];
  const int tid = threadIdx.x;
  if (tid < 64) { const int i = tid >> 3, j = tid & 7; a[tid] = (i == j) ? 0.f : alpha[tid]; }
  __syncthreads();
  const size_t g = (size_t)blockIdx.x * 256 + tid;    // over M_*DC_
  const int row = (int)(g / DC_);
  const int dc  = (int)(g % DC_);
  float vh[C_], hh[C_], uh[C_];
  #pragma unroll
  for (int c = 0; c < C_; ++c) {
    const size_t idx = ((size_t)c * M_ + row) * DC_ + dc;
    vh[c] = VH[idx]; hh[c] = H[idx]; uh[c] = UH[idx];
  }
  #pragma unroll
  for (int i = 0; i < C_; ++i) {
    float wv = 0.f;
    #pragma unroll
    for (int j = 0; j < C_; ++j) wv += a[i * 8 + j] * vh[j];
    const float m = hh[i] + uh[i] * wv;
    mOut[(size_t)row * D_ + i * DC_ + dc] = m;
    fuseH[(size_t)row * (2 * D_) + colBase + i * DC_ + dc] = (h16)m;
  }
}

__global__ void x1_kernel(const float* __restrict__ x, const float* __restrict__ beta,
                          const float* __restrict__ mf, const float* __restrict__ mb,
                          float* __restrict__ x1) {
  const size_t i = (size_t)blockIdx.x * 256 + threadIdx.x;
  const float bt = beta[i];
  x1[i] = x[i] + bt * mf[i] + (1.f - bt) * mb[i];
}

__global__ void fwdglob_kernel(const float* __restrict__ Hf, float* __restrict__ outG) {
  const int g2 = blockIdx.x * 256 + threadIdx.x;
  if (g2 < B_ * DC_) {
    const int b = g2 / DC_, dc = g2 % DC_;
    outG[g2] = Hf[((size_t)(C_ - 1) * M_ + (size_t)b * T_ + (T_ - 1)) * DC_ + dc];
  }
}

// ---------------------------------------------------------------------------
// Host orchestration
// ---------------------------------------------------------------------------
extern "C" void kernel_launch(void* const* d_in, const int* in_sizes, int n_in,
                              void* d_out, int out_size, void* d_ws, size_t ws_size,
                              hipStream_t stream) {
  (void)in_sizes; (void)n_in; (void)out_size;
  const float* x      = (const float*)d_in[0];
  const float* ln1_g  = (const float*)d_in[1];
  const float* ln1_b  = (const float*)d_in[2];
  const float* projW  = (const float*)d_in[3];
  const float* gfW    = (const float*)d_in[4];
  const float* gf_b   = (const float*)d_in[5];
  const float* gbW    = (const float*)d_in[6];
  const float* gb_b   = (const float*)d_in[7];
  const float* Uf     = (const float*)d_in[8];
  const float* Vf     = (const float*)d_in[9];
  const float* alphaF = (const float*)d_in[10];
  const float* Ub     = (const float*)d_in[11];
  const float* Vb     = (const float*)d_in[12];
  const float* alphaB = (const float*)d_in[13];
  const float* fuseW  = (const float*)d_in[14];
  const float* fuse_b = (const float*)d_in[15];
  const float* ln2_g  = (const float*)d_in[16];
  const float* ln2_b  = (const float*)d_in[17];
  const float* mlpW1  = (const float*)d_in[18];
  const float* mlp_b1 = (const float*)d_in[19];
  const float* mlpW2  = (const float*)d_in[20];
  const float* mlp_b2 = (const float*)d_in[21];

  float* out  = (float*)d_out;
  float* fwdg = out + (size_t)M_ * D_;

  const size_t MiB = 1ull << 20;
  if (ws_size < 432 * MiB) return;
  char* w = (char*)d_ws;

  // f16 weight arena (16 MiB)
  h16* projW16 = (h16*)w;
  h16* gfT  = projW16 + (size_t)C_ * DC_ * D_;
  h16* gbT  = gfT + (size_t)C_ * DC_ * DC_;
  h16* UfT  = gbT + (size_t)C_ * DC_ * DC_;
  h16* VfT  = UfT + (size_t)C_ * DC_ * DC_;
  h16* UbT  = VfT + (size_t)C_ * DC_ * DC_;
  h16* VbT  = UbT + (size_t)C_ * DC_ * DC_;
  h16* fuseW16 = VbT + (size_t)C_ * DC_ * DC_;
  h16* W1h = fuseW16 + (size_t)D_ * 2 * D_;
  h16* W2h = W1h + (size_t)MLP_ * D_;

  // Region-aliased activation slots
  char* A0 = w +  16 * MiB;  // 32MiB: xn_h -> Hfh -> Hbh -> x1n_h
  char* A1 = w +  48 * MiB;  // 64MiB: Zf32 -> Hf32 -> Hb32 -> beta
  char* A2 = w + 112 * MiB;  // 64MiB: Zh -> UHf -> UHb -> x1
  char* A3 = w + 176 * MiB;  // 64MiB: vf -> VHf -> VHb -> mlp_h
  char* A4 = w + 240 * MiB;  // 64MiB: vb -> mb32
  char* A5 = w + 304 * MiB;  // 64MiB: mf32
  char* A6 = w + 368 * MiB;  // 64MiB: fuse_in (f16, [M,2D])

  const size_t chS = (size_t)M_ * DC_;        // per-channel activation stride
  const size_t wS  = (size_t)DC_ * DC_;       // per-channel 128x128 weight stride

  // --- weight conversion ---
  cvt_kernel<<<4096, 256, 0, stream>>>(projW, projW16, (size_t)C_ * DC_ * D_);
  cvtT_kernel<<<C_ * DC_, DC_, 0, stream>>>(gfW, gfT);
  cvtT_kernel<<<C_ * DC_, DC_, 0, stream>>>(gbW, gbT);
  cvtT_kernel<<<C_ * DC_, DC_, 0, stream>>>(Uf,  UfT);
  cvtT_kernel<<<C_ * DC_, DC_, 0, stream>>>(Vf,  VfT);
  cvtT_kernel<<<C_ * DC_, DC_, 0, stream>>>(Ub,  UbT);
  cvtT_kernel<<<C_ * DC_, DC_, 0, stream>>>(Vb,  VbT);
  cvt_kernel<<<8192, 256, 0, stream>>>(fuseW, fuseW16, (size_t)D_ * 2 * D_);
  cvt_kernel<<<8192, 256, 0, stream>>>(mlpW1, W1h, (size_t)MLP_ * D_);
  cvt_kernel<<<8192, 256, 0, stream>>>(mlpW2, W2h, (size_t)D_ * MLP_);

  // --- LN1: x -> xn(f16) in A0 ---
  ln_kernel<<<M_, 256, 0, stream>>>(x, ln1_g, ln1_b, (h16*)A0);

  // --- Z[c] = xn @ projW[c]^T  -> Zf32(A1), Zh(A2) ---
  gemm_epi<1><<<dim3(DC_ / 64, M_ / 128, C_), 256, 0, stream>>>(
      (const h16*)A0, 0, projW16, (size_t)DC_ * D_, nullptr, 0, nullptr, 0,
      (float*)A1, chS, (h16*)A2, chS, M_, DC_, D_);

  // --- gated values: vf = sigmoid(Z@gfT + gf_b) * Z ; vb likewise ---
  gemm_epi<2><<<dim3(2, M_ / 128, C_), 256, 0, stream>>>(
      (const h16*)A2, chS, gfT, wS, gf_b, DC_, (const float*)A1, chS,
      (float*)A3, chS, nullptr, 0, M_, DC_, DC_);
  gemm_epi<2><<<dim3(2, M_ / 128, C_), 256, 0, stream>>>(
      (const h16*)A2, chS, gbT, wS, gb_b, DC_, (const float*)A1, chS,
      (float*)A4, chS, nullptr, 0, M_, DC_, DC_);

  // --- forward scan: vf -> Hf(A1), Hfh(A0); grab fwd_global ---
  scan_kernel<<<C_ * B_, DC_, 0, stream>>>((const float*)A3, (float*)A1, (h16*)A0, 0);
  fwdglob_kernel<<<2, 256, 0, stream>>>((const float*)A1, fwdg);

  // --- UHf(A2), VHf(A3); interfere -> mf(A5), fuse_in[:, :D](A6) ---
  gemm_epi<0><<<dim3(2, M_ / 128, C_), 256, 0, stream>>>(
      (const h16*)A0, chS, UfT, wS, nullptr, 0, nullptr, 0,
      (float*)A2, chS, nullptr, 0, M_, DC_, DC_);
  gemm_epi<0><<<dim3(2, M_ / 128, C_), 256, 0, stream>>>(
      (const h16*)A0, chS, VfT, wS, nullptr, 0, nullptr, 0,
      (float*)A3, chS, nullptr, 0, M_, DC_, DC_);
  interfere_kernel<<<(M_ * DC_) / 256, 256, 0, stream>>>(
      (const float*)A1, (const float*)A2, (const float*)A3, alphaF,
      (float*)A5, (h16*)A6, 0);

  // --- backward scan: vb -> Hb(A1), Hbh(A0) ---
  scan_kernel<<<C_ * B_, DC_, 0, stream>>>((const float*)A4, (float*)A1, (h16*)A0, 1);

  // --- UHb(A2), VHb(A3); interfere -> mb(A4), fuse_in[:, D:](A6) ---
  gemm_epi<0><<<dim3(2, M_ / 128, C_), 256, 0, stream>>>(
      (const h16*)A0, chS, UbT, wS, nullptr, 0, nullptr, 0,
      (float*)A2, chS, nullptr, 0, M_, DC_, DC_);
  gemm_epi<0><<<dim3(2, M_ / 128, C_), 256, 0, stream>>>(
      (const h16*)A0, chS, VbT, wS, nullptr, 0, nullptr, 0,
      (float*)A3, chS, nullptr, 0, M_, DC_, DC_);
  interfere_kernel<<<(M_ * DC_) / 256, 256, 0, stream>>>(
      (const float*)A1, (const float*)A2, (const float*)A3, alphaB,
      (float*)A4, (h16*)A6, D_);

  // --- beta = sigmoid([mf,mb] @ fuseW^T + fuse_b)  -> A1 ---
  gemm_epi<3><<<dim3(D_ / 64, M_ / 128, 1), 256, 0, stream>>>(
      (const h16*)A6, 0, fuseW16, 0, fuse_b, 0, nullptr, 0,
      (float*)A1, 0, nullptr, 0, M_, D_, 2 * D_);

  // --- x1 = x + beta*mf + (1-beta)*mb  -> A2 ---
  x1_kernel<<<(M_ * D_) / 256, 256, 0, stream>>>(
      x, (const float*)A1, (const float*)A5, (const float*)A4, (float*)A2);

  // --- LN2: x1 -> x1n(f16) in A0 ---
  ln_kernel<<<M_, 256, 0, stream>>>((const float*)A2, ln2_g, ln2_b, (h16*)A0);

  // --- MLP1: h = gelu(x1n @ W1^T + b1) -> f16 in A3 ---
  gemm_epi<4><<<dim3(MLP_ / 64, M_ / 128, 1), 256, 0, stream>>>(
      (const h16*)A0, 0, W1h, 0, mlp_b1, 0, nullptr, 0,
      nullptr, 0, (h16*)A3, 0, M_, MLP_, D_);

  // --- MLP2: out = x1 + h @ W2^T + b2 ---
  gemm_epi<5><<<dim3(D_ / 64, M_ / 128, 1), 256, 0, stream>>>(
      (const h16*)A3, 0, W2h, 0, mlp_b2, 0, (const float*)A2, 0,
      out, 0, nullptr, 0, M_, D_, MLP_);
}
